// MultiheadAttention_85263690760243
// MI455X (gfx1250) — matrix operand order repaired and checked by reading, not compiled, not run on hardware
//
#include <hip/hip_runtime.h>

// ---------------------------------------------------------------------------
// Agent-aware multihead attention for MI455X (gfx1250, wave32, WMMA).
// L=1024, N=8, E=512, H=8, head_dim=64, B=N*H=64.
// Flash-attention style: no materialized 64x1024x1024 logits.
// All matrix math via v_wmma_f32_16x16x32_f16 (fp32 accumulate).
// Dense GEMMs: 16x64 tile/wave, A-frag reused across 4 WMMAs, and operands
// software double-buffered so loads for step k+1 issue before WMMAs of step k.
// ---------------------------------------------------------------------------

typedef _Float16 v16h __attribute__((ext_vector_type(16)));
typedef _Float16 v8h  __attribute__((ext_vector_type(8)));
typedef float    v8f  __attribute__((ext_vector_type(8)));

#define L_  1024
#define N_  8
#define E_  512
#define H_  8
#define D_  64
#define B_  64      // N_*H_
#define M_  8192    // L_*N_
#define P5_ 2560    // 5*E_

__device__ __forceinline__ v8f wmma_f16(v16h a, v16h b, v8f c) {
    // (neg_a, A, neg_b, B, c_mod, C, reuse_a, reuse_b)
    return __builtin_amdgcn_wmma_f32_16x16x32_f16(false, a, false, b, (short)0, c,
                                                  false, false);
}

// A-matrix fragment (16x32 f16, row-major source, row stride ld elements).
// Lane layout: row = lane&15; lanes 0-15: K = {0..7,16..23}; lanes 16-31: +8.
__device__ __forceinline__ v16h load_afrag(const _Float16* base, int ld,
                                           int row0, int k0, int lr, int g) {
    const _Float16* p = base + (size_t)(row0 + lr) * ld + k0;
    union { v16h v; v8h h[2]; } u;
    u.h[0] = *(const v8h*)(p + g * 8);
    u.h[1] = *(const v8h*)(p + 16 + g * 8);
    return u.v;
}

// B-matrix fragment (32x16 f16) from Bt[n][k] row-major (n = output column).
// Lane layout: col = lane&15; lanes 0-15: K = {0..7,16..23}; lanes 16-31: +8.
__device__ __forceinline__ v16h load_bfrag(const _Float16* base, int ld,
                                           int n0, int k0, int lr, int g) {
    const _Float16* p = base + (size_t)(n0 + lr) * ld + k0;
    union { v16h v; v8h h[2]; } u;
    u.h[0] = *(const v8h*)(p + g * 8);
    u.h[1] = *(const v8h*)(p + 16 + g * 8);
    return u.v;
}

// ---------------------------------------------------------------------------
__global__ void cvt_f32_f16(const float* __restrict__ src,
                            _Float16* __restrict__ dst, int n) {
    int i = blockIdx.x * blockDim.x + threadIdx.x;
    int stride = gridDim.x * blockDim.x;
    for (; i < n; i += stride) dst[i] = (_Float16)src[i];
}

// Double-buffered 16x64 GEMM tile body: acc[t] += A(16xE) @ Bt(n0+t*16, :)^T.
__device__ __forceinline__ void gemm_16x64(const _Float16* __restrict__ Abase,
                                           const _Float16* __restrict__ Bbase,
                                           int m0, int n0, int lr, int g,
                                           v8f acc[4]) {
    v16h a0 = load_afrag(Abase, E_, m0, 0, lr, g);
    v16h b0[4];
#pragma unroll
    for (int t = 0; t < 4; ++t) b0[t] = load_bfrag(Bbase, E_, n0 + t * 16, 0, lr, g);
#pragma unroll
    for (int k0 = 0; k0 < E_; k0 += 32) {
        v16h a1 = {};
        v16h b1[4] = {{}, {}, {}, {}};
        if (k0 + 32 < E_) {
            a1 = load_afrag(Abase, E_, m0, k0 + 32, lr, g);
#pragma unroll
            for (int t = 0; t < 4; ++t)
                b1[t] = load_bfrag(Bbase, E_, n0 + t * 16, k0 + 32, lr, g);
        }
#pragma unroll
        for (int t = 0; t < 4; ++t) acc[t] = wmma_f16(a0, b0[t], acc[t]);
        a0 = a1;
#pragma unroll
        for (int t = 0; t < 4; ++t) b0[t] = b1[t];
    }
}

// ---------------------------------------------------------------------------
// proj = X(8192x512) @ W^T(512x2560) + bias, scattered into per-head fp16
// buffers: q*0.125, k, q_same*0.125, k_same as [b][l][64]; V as vT[b][d][l].
// One wave per 16x64 output tile. 64-wide tiles never cross a 512-chunk or
// 64-head boundary.
__global__ void proj_kernel(const _Float16* __restrict__ Xh,
                            const _Float16* __restrict__ Wh,
                            const float* __restrict__ bias,
                            _Float16* __restrict__ qb, _Float16* __restrict__ kb,
                            _Float16* __restrict__ vT, _Float16* __restrict__ qsb,
                            _Float16* __restrict__ ksb) {
    int wave = (blockIdx.x * blockDim.x + threadIdx.x) >> 5;
    int lane = threadIdx.x & 31;
    int lr = lane & 15, g = lane >> 4;
    const int ntiles = P5_ / 64;                 // 40
    int mt = wave / ntiles, nt = wave % ntiles;
    int m0 = mt * 16, n0 = nt * 64;

    v8f acc[4] = {{}, {}, {}, {}};
    gemm_16x64(Xh, Wh, m0, n0, lr, g, acc);

    int chunk = n0 >> 9;            // /512 : 0=q 1=k 2=v 3=q_same 4=k_same
    int h = (n0 & 511) >> 6;        // head (constant across the 64-wide tile)
    float scale = (chunk == 0 || chunk == 3) ? 0.125f : 1.0f;
    _Float16* dst = (chunk == 0) ? qb : (chunk == 1) ? kb
                  : (chunk == 2) ? vT : (chunk == 3) ? qsb : ksb;
#pragma unroll
    for (int t = 0; t < 4; ++t) {
        int d = t * 16 + lr;                     // 0..63 within the head
        float bj = bias[n0 + t * 16 + lr];
#pragma unroll
        for (int r = 0; r < 8; ++r) {
            int m = m0 + g * 8 + r;              // X row = l*8 + n
            int l = m >> 3, nb = m & 7;
            int b_ = nb * 8 + h;                 // attention batch = n*H + h
            float val = (acc[t][r] + bj) * scale;
            if (chunk == 2)
                dst[(((size_t)b_ * 64 + d) << 10) + l] = (_Float16)val;  // vT[b][d][l]
            else
                dst[(((size_t)b_ << 10) + l) * 64 + d] = (_Float16)val;  // [b][l][d]
        }
    }
}

// ---------------------------------------------------------------------------
// Flash attention: one wave per (b, 16-query-row tile). Online softmax with
// agent-aware mask blend. Row stats replicated per 16-lane group (matches the
// C-layout row ownership). P bounced through LDS to re-layout C->A.
// V fragments + mask values for a chunk are loaded at the top of the chunk so
// their latency overlaps the score WMMAs and the softmax VALU work.
__global__ void attn_kernel(const _Float16* __restrict__ q,
                            const _Float16* __restrict__ kbuf,
                            const _Float16* __restrict__ vT,
                            const _Float16* __restrict__ qs,
                            const _Float16* __restrict__ ksb,
                            const float* __restrict__ mask,
                            _Float16* __restrict__ attn16) {
    __shared__ _Float16 Pl[4][16][32];
    int waveId = threadIdx.x >> 5;
    int lane = threadIdx.x & 31;
    int lr = lane & 15, g = lane >> 4;
    int b = blockIdx.x;                       // 0..63
    int qt = blockIdx.y * 4 + waveId;         // 0..63
    int q0 = qt * 16;

    const _Float16* Qb  = q    + ((size_t)b << 10) * 64;   // [1024][64]
    const _Float16* Qsb = qs   + ((size_t)b << 10) * 64;
    const _Float16* Kb  = kbuf + ((size_t)b << 10) * 64;
    const _Float16* Ksb = ksb  + ((size_t)b << 10) * 64;
    const _Float16* Vb  = vT   + ((size_t)b << 6) * 1024;  // [64][1024]

    // Q fragments live in registers for the whole key loop (K=64 -> 2 frags each).
    v16h qa0  = load_afrag(Qb,  64, q0, 0,  lr, g);
    v16h qa1  = load_afrag(Qb,  64, q0, 32, lr, g);
    v16h qsa0 = load_afrag(Qsb, 64, q0, 0,  lr, g);
    v16h qsa1 = load_afrag(Qsb, 64, q0, 32, lr, g);

    v8f O0 = {}, O1 = {}, O2 = {}, O3 = {};   // 16x64 output accumulator (C-layout)
    float psum[8];
    float mrow[8];
#pragma unroll
    for (int r = 0; r < 8; ++r) { psum[r] = 0.0f; mrow[r] = -3.0e38f; }

    for (int s0 = 0; s0 < L_; s0 += 32) {
        // Issue chunk-invariant loads first: V fragments and mask values.
        v16h bv0 = load_bfrag(Vb, 1024, 0,  s0, lr, g);
        v16h bv1 = load_bfrag(Vb, 1024, 16, s0, lr, g);
        v16h bv2 = load_bfrag(Vb, 1024, 32, s0, lr, g);
        v16h bv3 = load_bfrag(Vb, 1024, 48, s0, lr, g);
        float mv[2][8];
#pragma unroll
        for (int sub = 0; sub < 2; ++sub)
#pragma unroll
            for (int r = 0; r < 8; ++r)
                mv[sub][r] = mask[(q0 + g * 8 + r) * 1024 + s0 + sub * 16 + lr];

        float S[2][8];
#pragma unroll
        for (int sub = 0; sub < 2; ++sub) {
            int n0 = s0 + sub * 16;
            v8f c = {};
            c = wmma_f16(qa0, load_bfrag(Kb, 64, n0, 0,  lr, g), c);
            c = wmma_f16(qa1, load_bfrag(Kb, 64, n0, 32, lr, g), c);
            v8f cs = {};
            cs = wmma_f16(qsa0, load_bfrag(Ksb, 64, n0, 0,  lr, g), cs);
            cs = wmma_f16(qsa1, load_bfrag(Ksb, 64, n0, 32, lr, g), cs);
#pragma unroll
            for (int r = 0; r < 8; ++r)
                S[sub][r] = c[r] + mv[sub][r] * (cs[r] - c[r]);  // diff + m*(same-diff)
        }
        // Online softmax update. Row (g*8+r) lives in the 16 lanes of group g.
        float scl[8];
#pragma unroll
        for (int r = 0; r < 8; ++r) {
            float v = fmaxf(S[0][r], S[1][r]);
            v = fmaxf(v, __shfl_xor(v, 1));
            v = fmaxf(v, __shfl_xor(v, 2));
            v = fmaxf(v, __shfl_xor(v, 4));
            v = fmaxf(v, __shfl_xor(v, 8));
            float mnew = fmaxf(mrow[r], v);
            scl[r] = __expf(mrow[r] - mnew);
            mrow[r] = mnew;
        }
#pragma unroll
        for (int r = 0; r < 8; ++r) {
            float pa = __expf(S[0][r] - mrow[r]);
            float pb = __expf(S[1][r] - mrow[r]);
            psum[r] = psum[r] * scl[r] + pa + pb;        // per-lane partial row sum
            Pl[waveId][g * 8 + r][lr]      = (_Float16)pa;
            Pl[waveId][g * 8 + r][16 + lr] = (_Float16)pb;
            O0[r] *= scl[r]; O1[r] *= scl[r]; O2[r] *= scl[r]; O3[r] *= scl[r];
        }
        // Same-wave DS ops are in-order; wait ensures cross-lane visibility.
        asm volatile("s_wait_dscnt 0" ::: "memory");
        v16h ap;
        {
            const _Float16* p = &Pl[waveId][lr][0];       // ld = 32
            union { v16h v; v8h h[2]; } u;
            u.h[0] = *(const v8h*)(p + g * 8);
            u.h[1] = *(const v8h*)(p + 16 + g * 8);
            ap = u.v;
        }
        // P(16x32) @ V(32x64): V fragments already in flight since loop top.
        O0 = wmma_f16(ap, bv0, O0);
        O1 = wmma_f16(ap, bv1, O1);
        O2 = wmma_f16(ap, bv2, O2);
        O3 = wmma_f16(ap, bv3, O3);
    }

    // Normalize and scatter to attn16[(l*8+n)][h*64+d] (fp16, out-proj A matrix).
    int nb = b >> 3, h = b & 7;
#pragma unroll
    for (int r = 0; r < 8; ++r) {
        float s = psum[r];
        s += __shfl_xor(s, 1);
        s += __shfl_xor(s, 2);
        s += __shfl_xor(s, 4);
        s += __shfl_xor(s, 8);
        float inv = 1.0f / s;
        int lq = q0 + g * 8 + r;
        size_t rowbase = ((size_t)lq * 8 + nb) * 512 + h * 64;
        attn16[rowbase +      lr] = (_Float16)(O0[r] * inv);
        attn16[rowbase + 16 + lr] = (_Float16)(O1[r] * inv);
        attn16[rowbase + 32 + lr] = (_Float16)(O2[r] * inv);
        attn16[rowbase + 48 + lr] = (_Float16)(O3[r] * inv);
    }
}

// ---------------------------------------------------------------------------
// out = attn16(8192x512) @ Wo^T + bias -> fp32 d_out (exactly (L,N,E) flat).
__global__ void outproj_kernel(const _Float16* __restrict__ A16,
                               const _Float16* __restrict__ Woh,
                               const float* __restrict__ bias,
                               float* __restrict__ out) {
    int wave = (blockIdx.x * blockDim.x + threadIdx.x) >> 5;
    int lane = threadIdx.x & 31;
    int lr = lane & 15, g = lane >> 4;
    const int ntiles = E_ / 64;                  // 8
    int mt = wave / ntiles, nt = wave % ntiles;
    int m0 = mt * 16, n0 = nt * 64;

    v8f acc[4] = {{}, {}, {}, {}};
    gemm_16x64(A16, Woh, m0, n0, lr, g, acc);

#pragma unroll
    for (int t = 0; t < 4; ++t) {
        float bj = bias[n0 + t * 16 + lr];
#pragma unroll
        for (int r = 0; r < 8; ++r) {
            int m = m0 + g * 8 + r;
            out[(size_t)m * 512 + n0 + t * 16 + lr] = acc[t][r] + bj;
        }
    }
}

// ---------------------------------------------------------------------------
extern "C" void kernel_launch(void* const* d_in, const int* in_sizes, int n_in,
                              void* d_out, int out_size, void* d_ws, size_t ws_size,
                              hipStream_t stream) {
    const float* query = (const float*)d_in[0];   // (L,N,E)
    const float* Wqkv  = (const float*)d_in[1];   // (5E,E)
    const float* bqkv  = (const float*)d_in[2];   // (5E,)
    const float* Wo    = (const float*)d_in[3];   // (E,E)
    const float* bo    = (const float*)d_in[4];   // (E,)
    const float* mask  = (const float*)d_in[5];   // (L,L)

    char* ws = (char*)d_ws;
    size_t off = 0;
    _Float16* Xh  = (_Float16*)(ws + off); off += (size_t)M_ * E_ * 2;   // 8 MB
    _Float16* Wh  = (_Float16*)(ws + off); off += (size_t)P5_ * E_ * 2;  // 2.5 MB
    _Float16* Woh = (_Float16*)(ws + off); off += (size_t)E_ * E_ * 2;   // 0.5 MB
    _Float16* qb  = (_Float16*)(ws + off); off += (size_t)B_ * L_ * D_ * 2;
    _Float16* kb  = (_Float16*)(ws + off); off += (size_t)B_ * L_ * D_ * 2;
    _Float16* qsb = (_Float16*)(ws + off); off += (size_t)B_ * L_ * D_ * 2;
    _Float16* ksb = (_Float16*)(ws + off); off += (size_t)B_ * L_ * D_ * 2;
    _Float16* vT  = (_Float16*)(ws + off); off += (size_t)B_ * L_ * D_ * 2;
    _Float16* attn16 = Xh;  // Xh is dead after proj_kernel -> reuse for attn output

    cvt_f32_f16<<<1024, 256, 0, stream>>>(query, Xh, M_ * E_);
    cvt_f32_f16<<<1024, 256, 0, stream>>>(Wqkv, Wh, P5_ * E_);
    cvt_f32_f16<<<512,  256, 0, stream>>>(Wo, Woh, E_ * E_);

    // 512 x 40 tiles (16x64 each), 4 waves/block -> exact grid (EXEC all-ones).
    proj_kernel<<<(M_ / 16) * (P5_ / 64) / 4, 128, 0, stream>>>(
        Xh, Wh, bqkv, qb, kb, vT, qsb, ksb);

    // 64 batches x 64 query tiles, 4 waves/block.
    attn_kernel<<<dim3(B_, (L_ / 16) / 4), 128, 0, stream>>>(
        qb, kb, vT, qsb, ksb, mask, attn16);

    // 512 x 8 tiles (16x64 each), 4 waves/block.
    outproj_kernel<<<(M_ / 16) * (E_ / 64) / 4, 128, 0, stream>>>(
        attn16, Woh, bo, (float*)d_out);
}
